// CausalSelfAttention_43138651521103
// MI455X (gfx1250) — matrix operand order repaired and checked by reading, not compiled, not run on hardware
//
#include <hip/hip_runtime.h>

typedef __bf16 bf16;
typedef __attribute__((ext_vector_type(16))) __bf16 v16bf;
typedef __attribute__((ext_vector_type(8)))  __bf16 v8bf;
typedef __attribute__((ext_vector_type(4)))  __bf16 v4bf;
typedef __attribute__((ext_vector_type(8)))  float  v8f;
typedef __attribute__((ext_vector_type(4)))  float  v4f;
typedef __attribute__((ext_vector_type(4)))  int    v4i;
typedef __attribute__((ext_vector_type(8)))  int    v8i;

typedef __attribute__((address_space(1))) v4i v4i_as1;   // global
typedef __attribute__((address_space(3))) v4i v4i_as3;   // LDS

#define TSEQ 2048
#define CDIM 1024
#define NH   16
#define DK   64
#define LDSS 40   // LDS row stride in bf16 elems (80 B) -> conflict-free b128 reads

#if defined(__has_builtin)
#if __has_builtin(__builtin_amdgcn_global_load_async_to_lds_b128)
#define HAS_ASYNC_LDS 1
#endif
#endif

// ---------- helpers ----------
static __device__ __forceinline__ bf16 f2b(float f) {
  unsigned u = __builtin_bit_cast(unsigned, f);
  u += 0x7FFFu + ((u >> 16) & 1u);               // round-to-nearest-even
  unsigned short s = (unsigned short)(u >> 16);
  return __builtin_bit_cast(bf16, s);
}

// pack two f32 -> packed bf16 pair in one u32 (low = a, high = b)
static __device__ __forceinline__ unsigned pack2bf(float a, float b) {
#if __has_builtin(__builtin_amdgcn_cvt_pk_bf16_f32)
  auto t = __builtin_amdgcn_cvt_pk_bf16_f32(a, b);
  return __builtin_bit_cast(unsigned, t);
#else
  unsigned short ua = __builtin_bit_cast(unsigned short, f2b(a));
  unsigned short ub = __builtin_bit_cast(unsigned short, f2b(b));
  return ((unsigned)ub << 16) | (unsigned)ua;
#endif
}

static __device__ __forceinline__ float fast_exp2(float x) {
#if __has_builtin(__builtin_amdgcn_exp2f)
  return __builtin_amdgcn_exp2f(x);
#else
  return __expf(x * 0.69314718055994531f);
#endif
}

static __device__ __forceinline__ float fast_rcp(float x) {
#if __has_builtin(__builtin_amdgcn_rcpf)
  return __builtin_amdgcn_rcpf(x);
#else
  return 1.0f / x;
#endif
}

static __device__ __forceinline__ v16bf cat8(v8bf lo, v8bf hi) {
  return __builtin_shufflevector(lo, hi, 0,1,2,3,4,5,6,7,8,9,10,11,12,13,14,15);
}

// A fragment (16x32 bf16): lane holds row (lane&15); 16 contiguous K at hi*16
static __device__ __forceinline__ v16bf load_a(const bf16* rowp, int hi) {
  v8bf lo = *(const v8bf*)(rowp + hi * 16);
  v8bf hh = *(const v8bf*)(rowp + hi * 16 + 8);
  return cat8(lo, hh);
}
// B fragment (32x16 bf16): lane holds 16 contiguous K elems of its source row
static __device__ __forceinline__ v16bf load_b(const bf16* p) {
  return cat8(*(const v8bf*)p, *(const v8bf*)(p + 8));
}

static __device__ __forceinline__ v8f wmma_bf16(v16bf a, v16bf b, v8f c) {
  return __builtin_amdgcn_wmma_f32_16x16x32_bf16(false, a, false, b, (short)0, c,
                                                 false, false);
}

// 16-byte async copy global -> LDS (CDNA5 GLOBAL_LOAD_ASYNC_TO_LDS_B128)
static __device__ __forceinline__ void async_cp16(bf16* lds, const bf16* g) {
#ifdef HAS_ASYNC_LDS
  __builtin_amdgcn_global_load_async_to_lds_b128((v4i_as1*)g, (v4i_as3*)lds, 0, 0);
#else
  *(v8bf*)lds = *(const v8bf*)g;
#endif
}
static __device__ __forceinline__ void wait_async_all() {
#ifdef HAS_ASYNC_LDS
#if __has_builtin(__builtin_amdgcn_s_wait_asynccnt)
  __builtin_amdgcn_s_wait_asynccnt(0);
#else
  asm volatile("s_wait_asynccnt 0x0" ::: "memory");
#endif
#endif
}

// ---------- kernel 0: f32 -> bf16 convert ----------
__global__ __launch_bounds__(256) void cvt_kernel(const float* __restrict__ src,
                                                  bf16* __restrict__ dst, int n4) {
  int i = blockIdx.x * blockDim.x + threadIdx.x;
  if (i < n4) {
    v4f v = ((const v4f*)src)[i];
    v4bf p = {f2b(v.x), f2b(v.y), f2b(v.z), f2b(v.w)};
    ((v4bf*)dst)[i] = p;
  }
}

// ---------- GEMM: C = A(bf16)[M,K] * B(bf16)[N,K]^T, async double-buffered ----------
// MODE 0: QKV epilogue (bias + scatter to Q/K/Vt bf16);  MODE 1: f32 out + bias
template <int MODE>
__global__ __launch_bounds__(256) void gemm_bf16_kernel(
    const bf16* __restrict__ A, const bf16* __restrict__ Bm,
    const float* __restrict__ bias, bf16* __restrict__ Qb,
    bf16* __restrict__ Kb, bf16* __restrict__ Vt, float* __restrict__ Out,
    int Kdim) {
  __shared__ alignas(16) bf16 sA[2][128 * LDSS];
  __shared__ alignas(16) bf16 sB[2][128 * LDSS];
  const int tid = threadIdx.x;
  const int lane = tid & 31, wave = tid >> 5;
  const int wm = wave & 3, wn = wave >> 2;      // 4x2 waves: 32x64 tile per wave
  const int rlo = lane & 15, hi = lane >> 4;
  const int m0 = blockIdx.y * 128;
  const int n0 = blockIdx.x * 128;
  const int srow = tid >> 2, scnk = tid & 3;    // staging: 4x16B chunks per 32-elem row
  v8f acc[2][4] = {};

  auto stage = [&](int buf, int k0) {
#pragma unroll
    for (int i = 0; i < 2; ++i) {
      int row = srow + i * 64;
      async_cp16(&sA[buf][row * LDSS + scnk * 8],
                 A + (size_t)(m0 + row) * Kdim + k0 + scnk * 8);
      async_cp16(&sB[buf][row * LDSS + scnk * 8],
                 Bm + (size_t)(n0 + row) * Kdim + k0 + scnk * 8);
    }
  };

  const int nit = Kdim >> 5;
  stage(0, 0);
  for (int it = 0; it < nit; ++it) {
    const int cur = it & 1;
    wait_async_all();            // own-wave async writes into buf[cur] done
    __syncthreads();             // everyone's writes visible / reads of buf[cur^1] done
    if (it + 1 < nit) stage(cur ^ 1, (it + 1) * 32);  // DMA overlaps WMMA below
    v16bf af[2], bfr[4];
#pragma unroll
    for (int mt = 0; mt < 2; ++mt)
      af[mt] = load_a(&sA[cur][(wm * 32 + mt * 16 + rlo) * LDSS], hi);
#pragma unroll
    for (int nt = 0; nt < 4; ++nt)
      bfr[nt] = load_b(&sB[cur][(wn * 64 + nt * 16 + rlo) * LDSS + hi * 16]);
#pragma unroll
    for (int mt = 0; mt < 2; ++mt)
#pragma unroll
      for (int nt = 0; nt < 4; ++nt)
        acc[mt][nt] = wmma_bf16(af[mt], bfr[nt], acc[mt][nt]);
  }

#pragma unroll
  for (int nt = 0; nt < 4; ++nt) {
    int o = n0 + wn * 64 + nt * 16 + rlo;
    float bv = bias[o];
    if constexpr (MODE == 0) {
      int h = o / 192;
      int rem = o - h * 192;
      int which = rem >> 6, d = rem & 63;
#pragma unroll
      for (int mt = 0; mt < 2; ++mt)
#pragma unroll
        for (int r = 0; r < 8; ++r) {
          int m = m0 + wm * 32 + mt * 16 + r + 8 * hi;
          int b = m >> 11, t = m & (TSEQ - 1);
          bf16 val = f2b(acc[mt][nt][r] + bv);
          size_t bh = (size_t)(b * NH + h);
          if (which == 0)      Qb[(bh * TSEQ + t) * DK + d] = val;
          else if (which == 1) Kb[(bh * TSEQ + t) * DK + d] = val;
          else                 Vt[(bh * DK + d) * TSEQ + t] = val;
        }
    } else {
#pragma unroll
      for (int mt = 0; mt < 2; ++mt)
#pragma unroll
        for (int r = 0; r < 8; ++r) {
          int m = m0 + wm * 32 + mt * 16 + r + 8 * hi;
          Out[(size_t)m * CDIM + o] = acc[mt][nt][r] + bv;
        }
    }
  }
}

// ---------- kernel 2: flash attention, transposed-score formulation ----------
// S^T tile = K_tile(16s x 64d) x Q^T;  lane holds one query column t = qt + (lane&15).
// ctx^T = V^T x P^T accumulated in registers; no LDS at all.
__global__ __launch_bounds__(256) void attn_kernel(
    const bf16* __restrict__ Q, const bf16* __restrict__ K,
    const bf16* __restrict__ Vt, bf16* __restrict__ Ctx) {
  const int tid = threadIdx.x;
  const int lane = tid & 31, wave = tid >> 5;
  const int rlo = lane & 15, hi = lane >> 4;
  // qt is wave-uniform; pin it to an SGPR so loop bound / tile-1 guard are scalar
  // branches and EXEC provably stays all-ones around every WMMA.
  const int qt = __builtin_amdgcn_readfirstlane((blockIdx.x * 8 + wave) * 16);
  const int bh = blockIdx.y;
  const int b = bh >> 4, h = bh & 15;
  const bf16* Qp = Q + (size_t)bh * TSEQ * DK;
  const bf16* Kp = K + (size_t)bh * TSEQ * DK;
  const bf16* Vp = Vt + (size_t)bh * DK * TSEQ;

  // Q as B-operand (d is the WMMA K-dim): lane holds 16 contiguous d of row qt+rlo
  const bf16* qrow = Qp + (size_t)(qt + rlo) * DK + hi * 16;
  v16bf bq0 = load_b(qrow);        // d = 0..31 (split across lane halves)
  v16bf bq1 = load_b(qrow + 32);   // d = 32..63

  const float SCL = 0.18033688011112042f;   // (1/8) * log2(e); exp2 domain
  const int tcol = qt + rlo;                // this lane's query index
  float mrun = -1e30f, lrun = 0.f;
  v8f acc[4] = {};                          // ctx^T tiles: rows d = dt*16 + r + 8*hi

  const int send = qt + 16;                 // exclusive causal bound (scalar)
  for (int s0 = 0; s0 < send; s0 += 32) {
    v8f sc0 = {}, sc1 = {};                 // S^T tiles: row s = s0(+16) + r + 8*hi
    {
      const bf16* krow = Kp + (size_t)(s0 + rlo) * DK;
      sc0 = wmma_bf16(load_a(krow, hi), bq0, sc0);
      sc0 = wmma_bf16(load_a(krow + 32, hi), bq1, sc0);
    }
    if (s0 + 16 < send) {                   // scalar guard (qt, s0 uniform)
      const bf16* krow = Kp + (size_t)(s0 + 16 + rlo) * DK;
      sc1 = wmma_bf16(load_a(krow, hi), bq0, sc1);
      sc1 = wmma_bf16(load_a(krow + 32, hi), bq1, sc1);
    }
    // V^T A-fragments (hoisted; global loads overlap softmax VALU)
    v16bf vfrag[4];
#pragma unroll
    for (int dt = 0; dt < 4; ++dt)
      vfrag[dt] = load_a(Vp + (size_t)(dt * 16 + rlo) * TSEQ + s0, hi);

    // masked, scaled scores for this lane's column
    float x[16];
#pragma unroll
    for (int r = 0; r < 8; ++r) {
      x[r]     = (s0 + r + 8 * hi      <= tcol) ? sc0[r] * SCL : -1e30f;
      x[8 + r] = (s0 + 16 + r + 8 * hi <= tcol) ? sc1[r] * SCL : -1e30f;
    }
    // in-lane max + one cross-half merge
    float mv = x[0];
#pragma unroll
    for (int e = 1; e < 16; ++e) mv = fmaxf(mv, x[e]);
    mv = fmaxf(mv, __shfl_xor(mv, 16));
    float mn = fmaxf(mrun, mv);
    float al = fast_exp2(mrun - mn);
    float p[16];
#pragma unroll
    for (int e = 0; e < 16; ++e) p[e] = fast_exp2(x[e] - mn);
    float rs = p[0];
#pragma unroll
    for (int e = 1; e < 16; ++e) rs += p[e];
    rs += __shfl_xor(rs, 16);
    lrun = lrun * al + rs;
    mrun = mn;
#pragma unroll
    for (int dt = 0; dt < 4; ++dt)
#pragma unroll
      for (int r = 0; r < 8; ++r) acc[dt][r] *= al;

    // Build P^T B-operand in registers: lane needs s-range (hi?16:0)+0..15 of col t.
    // Own lane half holds one 8-s span; partner (xor 16) holds the other.
    unsigned pk0[4], pk1[4], sw0[4], sw1[4];
#pragma unroll
    for (int j = 0; j < 4; ++j) {
      pk0[j] = pack2bf(p[2 * j], p[2 * j + 1]);
      pk1[j] = pack2bf(p[8 + 2 * j], p[9 + 2 * j]);
    }
#pragma unroll
    for (int j = 0; j < 4; ++j) {
      sw0[j] = __shfl_xor(pk0[j], 16);
      sw1[j] = __shfl_xor(pk1[j], 16);
    }
    v8i w;
#pragma unroll
    for (int j = 0; j < 4; ++j) {
      w[j]     = (int)(hi ? sw1[j] : pk0[j]);   // k = (hi?16:0)+0..7
      w[4 + j] = (int)(hi ? pk1[j] : sw0[j]);   // k = (hi?16:0)+8..15
    }
    v16bf bp = __builtin_bit_cast(v16bf, w);
#pragma unroll
    for (int dt = 0; dt < 4; ++dt)
      acc[dt] = wmma_bf16(vfrag[dt], bp, acc[dt]);
  }

  // normalize + vectorized store: lane holds 8 consecutive d per tile for query tcol
  float rinv = fast_rcp(lrun);
  bf16* cbase = Ctx + ((size_t)b * TSEQ + tcol) * CDIM + h * DK + 8 * hi;
#pragma unroll
  for (int dt = 0; dt < 4; ++dt) {
    v4i pk;
#pragma unroll
    for (int j = 0; j < 4; ++j)
      pk[j] = (int)pack2bf(acc[dt][2 * j] * rinv, acc[dt][2 * j + 1] * rinv);
    *(v8bf*)(cbase + dt * 16) = __builtin_bit_cast(v8bf, pk);
  }
}

// ---------- launch ----------
extern "C" void kernel_launch(void* const* d_in, const int* in_sizes, int n_in,
                              void* d_out, int out_size, void* d_ws, size_t ws_size,
                              hipStream_t stream) {
  const float* x    = (const float*)d_in[0];
  const float* Wqkv = (const float*)d_in[1];
  const float* bqkv = (const float*)d_in[2];
  const float* Wout = (const float*)d_in[3];
  const float* bout = (const float*)d_in[4];
  float* out = (float*)d_out;

  const size_t XE  = (size_t)4 * TSEQ * CDIM;   // 8,388,608 elems
  const size_t WQE = (size_t)3 * CDIM * CDIM;   // 3,145,728
  const size_t WOE = (size_t)CDIM * CDIM;       // 1,048,576

  bf16* Xb    = (bf16*)d_ws;       // x in bf16; reused as Ctx after QKV GEMM
  bf16* Wqkvb = Xb + XE;
  bf16* Woutb = Wqkvb + WQE;
  bf16* Qb    = Woutb + WOE;
  bf16* Kb    = Qb + XE;
  bf16* Vt    = Kb + XE;           // V transposed [B,H,dk,T]
  bf16* Ctx   = Xb;                // alias: x is dead after kernel 1

  cvt_kernel<<<(int)(XE / 1024),  256, 0, stream>>>(x,    Xb,    (int)(XE / 4));
  cvt_kernel<<<(int)(WQE / 1024), 256, 0, stream>>>(Wqkv, Wqkvb, (int)(WQE / 4));
  cvt_kernel<<<(int)(WOE / 1024), 256, 0, stream>>>(Wout, Woutb, (int)(WOE / 4));

  // QKV projection: M=8192, N=3072, K=1024
  gemm_bf16_kernel<0><<<dim3(3072 / 128, 8192 / 128), 256, 0, stream>>>(
      Xb, Wqkvb, bqkv, Qb, Kb, Vt, nullptr, CDIM);
  // Flash attention
  attn_kernel<<<dim3(TSEQ / 128, 4 * NH), 256, 0, stream>>>(Qb, Kb, Vt, Ctx);
  // Output projection: M=8192, N=1024, K=1024
  gemm_bf16_kernel<1><<<dim3(1024 / 128, 8192 / 128), 256, 0, stream>>>(
      Ctx, Woutb, bout, nullptr, nullptr, nullptr, out, CDIM);
}
